// _ProtoAggAttentionLayer_29463475651137
// MI455X (gfx1250) — matrix operand order repaired and checked by reading, not compiled, not run on hardware
//
#include <hip/hip_runtime.h>

// ProtoAggAttentionLayer, algebraically reduced:
//   softmax over the query axis followed by a sum over the query axis makes
//   the attention weights sum to 1 per (b,k), so Q/K/att/softmax cancel:
//   out[b,e] = sum_p sum_d x[b,p,d] * Wv[p,e,d]  +  sum_p bv[p,e]
// => single GEMM  [4096 x 16384] @ [16384 x 512]  (K-major on both sides)
// computed with v_wmma_f32_16x16x32_bf16 using a bf16 hi/lo split of both
// operands (3 cross-term WMMAs) for ~f32 accuracy with f32 accumulation.

typedef __attribute__((ext_vector_type(16))) __bf16       v16bf;
typedef __attribute__((ext_vector_type(8)))  float        v8f;
typedef __attribute__((ext_vector_type(4)))  float        v4f;
typedef __attribute__((ext_vector_type(4)))  unsigned int v4u;
typedef __attribute__((ext_vector_type(2)))  unsigned int v2u;

#define BATCH  4096
#define NPROT  32
#define DIM    512
#define KTOT   (NPROT * DIM)   // 16384
#define MT     128             // C-tile rows per block
#define NT     128             // C-tile cols per block
#define KT     64              // K-tile per LDS stage
#define LDSS   72              // padded LDS row stride (bf16 elements)

__device__ __forceinline__ unsigned int bf16_rne(float f) {
  unsigned int u = __float_as_uint(f);
  u += 0x7FFFu + ((u >> 16) & 1u);
  return u >> 16;
}
__device__ __forceinline__ float bf16_f(unsigned int h) {
  return __uint_as_float(h << 16);
}

// Split 4 f32 into bf16 hi/lo parts and store 8B each to LDS.
__device__ __forceinline__ void split_store4(unsigned short* hi, unsigned short* lo,
                                             int idx, v4f v) {
  unsigned int h[4], l[4];
#pragma unroll
  for (int i = 0; i < 4; ++i) {
    h[i] = bf16_rne(v[i]);
    l[i] = bf16_rne(v[i] - bf16_f(h[i]));
  }
  v2u hp = { h[0] | (h[1] << 16), h[2] | (h[3] << 16) };
  v2u lp = { l[0] | (l[1] << 16), l[2] | (l[3] << 16) };
  *(v2u*)(hi + idx) = hp;
  *(v2u*)(lo + idx) = lp;
}

union FragU { v16bf v; v4u q[2]; };

// A fragment (16x32 bf16): lane half ks: elems 0..7 = K ks*8+[0,8),
// elems 8..15 = K 16+ks*8+[0,8)   (ISA 16-bit A-matrix layout)
__device__ __forceinline__ v16bf load_frag_a(const unsigned short* lds, int row,
                                             int kk, int lane) {
  const int m = lane & 15, ks = lane >> 4;
  const unsigned short* p = lds + (row + m) * LDSS + kk + ks * 8;
  FragU u;
  u.q[0] = *(const v4u*)(p);
  u.q[1] = *(const v4u*)(p + 16);
  return u.v;
}

// B fragment from B^T (k-major rows): lane half ks: elems 0..7 = K ks*8+[0,8),
// elems 8..15 = K 16+ks*8+[0,8)   (same K order as A fragment)
__device__ __forceinline__ v16bf load_frag_b(const unsigned short* lds, int col,
                                             int kk, int lane) {
  const int n = lane & 15, ks = lane >> 4;
  const unsigned short* p = lds + (col + n) * LDSS + kk + ks * 8;
  FragU u;
  u.q[0] = *(const v4u*)(p);
  u.q[1] = *(const v4u*)(p + 16);
  return u.v;
}

__global__ __launch_bounds__(256, 1)
void proto_v_gemm_kernel(const float* __restrict__ x, const float* __restrict__ Wv,
                         const float* __restrict__ bv, float* __restrict__ out) {
  __shared__ unsigned short Ah[MT * LDSS];
  __shared__ unsigned short Al[MT * LDSS];
  __shared__ unsigned short Bh[NT * LDSS];
  __shared__ unsigned short Bl[NT * LDSS];

  const int t    = threadIdx.x;
  const int lane = t & 31;
  const int w    = t >> 5;
  const int wr   = w & 3;                // wave row quadrant -> rows wr*32..+31
  const int wc   = w >> 2;               // wave col half     -> cols wc*64..+63
  const int m0   = blockIdx.y * MT;
  const int n0   = blockIdx.x * NT;

  v8f acc[2][4];
#pragma unroll
  for (int i = 0; i < 2; ++i)
#pragma unroll
    for (int j = 0; j < 4; ++j)
#pragma unroll
      for (int e = 0; e < 8; ++e) acc[i][j][e] = 0.0f;

  for (int k0 = 0; k0 < KTOT; k0 += KT) {
    const int p  = k0 >> 9;              // proto index (KT=64 divides DIM=512)
    const int d0 = k0 & (DIM - 1);

    __syncthreads();                     // previous stage fully consumed
#pragma unroll
    for (int c = 0; c < 8; ++c) {
      const int flat = t + 256 * c;      // 2048 float4 loads per tile pair
      const int row  = flat >> 4;        // 0..127
      const int col  = (flat & 15) << 2; // 0,4,...,60
      v4f a = *(const v4f*)(x + (m0 + row) * KTOT + k0 + col);
      split_store4(Ah, Al, row * LDSS + col, a);
      v4f b = *(const v4f*)(Wv + p * (DIM * DIM) + (n0 + row) * DIM + d0 + col);
      split_store4(Bh, Bl, row * LDSS + col, b);
    }
    __syncthreads();

#pragma unroll
    for (int kk = 0; kk < KT; kk += 32) {
      v16bf ah[2], al[2], bh[4], bl[4];
#pragma unroll
      for (int i = 0; i < 2; ++i) {
        ah[i] = load_frag_a(Ah, wr * 32 + 16 * i, kk, lane);
        al[i] = load_frag_a(Al, wr * 32 + 16 * i, kk, lane);
      }
#pragma unroll
      for (int j = 0; j < 4; ++j) {
        bh[j] = load_frag_b(Bh, wc * 64 + 16 * j, kk, lane);
        bl[j] = load_frag_b(Bl, wc * 64 + 16 * j, kk, lane);
      }
#pragma unroll
      for (int i = 0; i < 2; ++i)
#pragma unroll
        for (int j = 0; j < 4; ++j) {
          acc[i][j] = __builtin_amdgcn_wmma_f32_16x16x32_bf16(
              false, ah[i], false, bh[j], (short)0, acc[i][j], false, false);
          acc[i][j] = __builtin_amdgcn_wmma_f32_16x16x32_bf16(
              false, ah[i], false, bl[j], (short)0, acc[i][j], false, false);
          acc[i][j] = __builtin_amdgcn_wmma_f32_16x16x32_bf16(
              false, al[i], false, bh[j], (short)0, acc[i][j], false, false);
        }
    }
  }

  // bias row: bsum[e] = sum_p bv[p, e]  (bv is 64 KB -> L2 resident)
  float bias[4];
#pragma unroll
  for (int j = 0; j < 4; ++j) {
    const int col = n0 + wc * 64 + 16 * j + (lane & 15);
    float s = 0.0f;
    for (int pp = 0; pp < NPROT; ++pp) s += bv[pp * DIM + col];
    bias[j] = s;
  }

  // C/D layout: vgpr v, lane L -> row = v + 8*(L>>4), col = L&15
#pragma unroll
  for (int i = 0; i < 2; ++i) {
    const int rbase = m0 + wr * 32 + 16 * i + (lane >> 4) * 8;
#pragma unroll
    for (int j = 0; j < 4; ++j) {
      const int col = n0 + wc * 64 + 16 * j + (lane & 15);
#pragma unroll
      for (int v = 0; v < 8; ++v) {
        out[(rbase + v) * DIM + col] = acc[i][j][v] + bias[j];
      }
    }
  }
}

extern "C" void kernel_launch(void* const* d_in, const int* in_sizes, int n_in,
                              void* d_out, int out_size, void* d_ws, size_t ws_size,
                              hipStream_t stream) {
  (void)in_sizes; (void)n_in; (void)d_ws; (void)ws_size; (void)out_size;
  // setup_inputs order: x, Wq, bq, Wk, bk, Wv, bv  (only x, Wv, bv are needed)
  const float* x  = (const float*)d_in[0];
  const float* Wv = (const float*)d_in[5];
  const float* bv = (const float*)d_in[6];
  float* out = (float*)d_out;

  dim3 grid(DIM / NT, BATCH / MT);   // (4, 32): n-blocks adjacent -> x shared via L2
  dim3 block(256);                   // 8 wave32 waves
  hipLaunchKernelGGL(proto_v_gemm_kernel, grid, block, 0, stream, x, Wv, bv, out);
}
